// SVD_Weights_28570122453321
// MI455X (gfx1250) — matrix operand order repaired and checked by reading, not compiled, not run on hardware
//
#include <hip/hip_runtime.h>
#include <hip/hip_bf16.h>

typedef __attribute__((ext_vector_type(16))) __bf16 v16bf;
typedef __attribute__((ext_vector_type(8)))  float  v8f;

#define NF_K   256     // K (NF)
#define IN_N   2048    // N (INPUT_SIZE)
#define GATE_M 8192    // NUM_GATE*OUTPUT_SIZE
#define OUT_M  10240   // GATE_M + OUTPUT_SIZE

// ---------------- kernel 1: s[256] = weight_b @ tag ----------------
__global__ void svd_scale_kernel(const float* __restrict__ wb,   // [256,20]
                                 const float* __restrict__ tag,  // [20,1]
                                 float* __restrict__ s) {
  const int t = threadIdx.x;       // 256 threads, one per NF row
  float acc = 0.f;
#pragma unroll
  for (int i = 0; i < 20; ++i) acc += wb[t * 20 + i] * tag[i];
  s[t] = acc;
}

// helper: convert 8 scaled f32 -> bf16 fragment slots [off..off+7]
__device__ __forceinline__ void cvt8(v16bf& d, int off, float4 p0, float4 p1,
                                     const float* __restrict__ sp) {
  d[off + 0] = (__bf16)(p0.x * sp[0]);
  d[off + 1] = (__bf16)(p0.y * sp[1]);
  d[off + 2] = (__bf16)(p0.z * sp[2]);
  d[off + 3] = (__bf16)(p0.w * sp[3]);
  d[off + 4] = (__bf16)(p1.x * sp[4]);
  d[off + 5] = (__bf16)(p1.y * sp[5]);
  d[off + 6] = (__bf16)(p1.z * sp[6]);
  d[off + 7] = (__bf16)(p1.w * sp[7]);
}

// ---------------- kernel 2: shared_w = (A * s) @ B -----------------
// A: weight_a [8192,256] f32, B: weight_c [256,2048] f32, out region [8192,2048]
__global__ __launch_bounds__(256)
void svd_gemm_wmma_kernel(const float* __restrict__ A,
                          const float* __restrict__ B,
                          const float* __restrict__ s,
                          float* __restrict__ out) {
  __shared__ float sSm[NF_K];
  sSm[threadIdx.x] = s[threadIdx.x];
  __syncthreads();

  const int lane = threadIdx.x & 31;
  const int wave = threadIdx.x >> 5;
  const int lrow = lane & 15;      // row (A), col (B/D) within 16x16 tile
  const int lhi  = lane >> 4;      // half-wave selector
  const int wm   = wave & 1;       // 2 waves along M  -> 64 rows each
  const int wn   = wave >> 1;      // 4 waves along N  -> 32 cols each
  const int rowBase = blockIdx.y * 128 + wm * 64;
  const int colBase = blockIdx.x * 128 + wn * 32;

  v8f acc[4][2] = {};

  for (int kk = 0; kk < NF_K; kk += 32) {
    // ---- A fragments: 16x32 bf16, scaled by s[k] from LDS ----
    const int kbA = kk + lhi * 8;
    const float* __restrict__ sp = &sSm[kbA];
    v16bf a[4];
#pragma unroll
    for (int mi = 0; mi < 4; ++mi) {
      const float* ap = A + (size_t)(rowBase + 16 * mi + lrow) * NF_K + kbA;
      float4 x0 = *(const float4*)(ap + 0);
      float4 x1 = *(const float4*)(ap + 4);
      float4 y0 = *(const float4*)(ap + 16);
      float4 y1 = *(const float4*)(ap + 20);
      cvt8(a[mi], 0, x0, x1, sp);
      cvt8(a[mi], 8, y0, y1, sp + 16);
    }

    // ---- B fragments: 32x16 bf16, column gather from weight_c ----
    const int kbB = kk + lhi * 8;
    v16bf b[2];
#pragma unroll
    for (int ni = 0; ni < 2; ++ni) {
      const float* bp = B + (size_t)kbB * IN_N + (colBase + 16 * ni + lrow);
#pragma unroll
      for (int j = 0; j < 16; ++j) b[ni][j] = (__bf16)bp[(size_t)(j + (j & 8)) * IN_N];
    }

    // ---- 8 WMMAs per K-step ----
#pragma unroll
    for (int mi = 0; mi < 4; ++mi)
#pragma unroll
      for (int ni = 0; ni < 2; ++ni)
        acc[mi][ni] = __builtin_amdgcn_wmma_f32_16x16x32_bf16(
            false, a[mi], false, b[ni], (short)0, acc[mi][ni], false, false);
  }

  // ---- store D fragments (coalesced 64B half-wave rows) ----
#pragma unroll
  for (int mi = 0; mi < 4; ++mi) {
#pragma unroll
    for (int ni = 0; ni < 2; ++ni) {
      const int col = colBase + 16 * ni + lrow;
#pragma unroll
      for (int r = 0; r < 8; ++r) {
        const int row = rowBase + 16 * mi + lhi * 8 + r;
        out[(size_t)row * IN_N + col] = acc[mi][ni][r];
      }
    }
  }
}

// ---------------- kernel 3: concat copies --------------------------
__global__ void svd_copy_f4_kernel(const float4* __restrict__ src,
                                   float4* __restrict__ dst, int n) {
  int i = blockIdx.x * blockDim.x + threadIdx.x;
  if (i < n) dst[i] = src[i];
}

__global__ void svd_copy_bias_kernel(const float* __restrict__ bias,
                                     const float* __restrict__ ibias,
                                     float* __restrict__ dst) {
  int i = blockIdx.x * blockDim.x + threadIdx.x;
  if (i < GATE_M)      dst[i] = bias[i];
  else if (i < OUT_M)  dst[i] = ibias[i - GATE_M];
}

extern "C" void kernel_launch(void* const* d_in, const int* in_sizes, int n_in,
                              void* d_out, int out_size, void* d_ws, size_t ws_size,
                              hipStream_t stream) {
  const float* tag  = (const float*)d_in[0];  // [20,1]
  const float* wa   = (const float*)d_in[1];  // [8192,256]
  const float* wb   = (const float*)d_in[2];  // [256,20]
  const float* wc   = (const float*)d_in[3];  // [256,2048]
  const float* bias = (const float*)d_in[4];  // [8192]
  const float* iw   = (const float*)d_in[5];  // [2048,2048]
  const float* ib   = (const float*)d_in[6];  // [2048]
  float* out = (float*)d_out;
  float* sv  = (float*)d_ws;                  // 256 floats of scratch

  // 1) s = weight_b @ tag
  svd_scale_kernel<<<1, 256, 0, stream>>>(wb, tag, sv);

  // 2) shared_w GEMM via bf16 WMMA, f32 accumulate
  dim3 grid(IN_N / 128, GATE_M / 128);        // 16 x 64 blocks
  svd_gemm_wmma_kernel<<<grid, 256, 0, stream>>>(wa, wc, sv, out);

  // 3) concat individual_weight + biases
  const int n4 = (2048 * 2048) / 4;
  svd_copy_f4_kernel<<<(n4 + 255) / 256, 256, 0, stream>>>(
      (const float4*)iw, (float4*)(out + (size_t)GATE_M * IN_N), n4);
  svd_copy_bias_kernel<<<(OUT_M + 255) / 256, 256, 0, stream>>>(
      bias, ib, out + (size_t)OUT_M * IN_N);
}
